// CustomMultiheadAttention_1254130451107
// MI455X (gfx1250) — matrix operand order repaired and checked by reading, not compiled, not run on hardware
//
#include <hip/hip_runtime.h>
#include <hip/hip_bf16.h>

// MI455X / gfx1250 causal multi-head attention, WMMA f16 (f32 accum).
// Compute-bound (137 GFLOP vs ~9us of HBM traffic at 23.3 TB/s): all matmuls
// via v_wmma_f32_16x16x32_f16. Flash-style online softmax; scores computed
// TRANSPOSED (S^T = K*Q^T) so each lane owns one query row -> row max/sum are
// in-lane trees + a single shfl_xor(16), and the output is accumulated as
// O^T = V^T * P^T so the softmax correction is a per-lane scalar. Causal mask
// applied only on the (uniform-branch) diagonal-straddling blocks.
// attn_map from a bounded recompute pass using saved row max/sum.

typedef __attribute__((ext_vector_type(16))) _Float16 v16h;
typedef __attribute__((ext_vector_type(8)))  _Float16 v8h;
typedef __attribute__((ext_vector_type(8)))  float    v8f;

union AF { v16h v; v8h h[2]; };

constexpr int S_ = 2048, B_ = 4, E_ = 1024, H_ = 16, D_ = 64;
constexpr int MROWS = S_ * B_;  // 8192 rows, row index r = s*B + b

__device__ __forceinline__ v8f zero8() { v8f z = {0.f,0.f,0.f,0.f,0.f,0.f,0.f,0.f}; return z; }

__device__ __forceinline__ v8f wmma_ff(AF a, AF b, v8f c) {
  // D = A(16x32 f16) * B(32x16 f16) + C(16x16 f32)
  return __builtin_amdgcn_wmma_f32_16x16x32_f16(false, a.v, false, b.v, (short)0, c, false, false);
}

// ---------------------------------------------------------------- convert
__global__ void cvt_f32_f16_x4(const float* __restrict__ s, _Float16* __restrict__ d, int n4) {
  int i = blockIdx.x * 256 + threadIdx.x;
  if (i < n4) {
    float4 f = ((const float4*)s)[i];
    union { _Float16 h[4]; uint2 u; } p;
    p.h[0] = (_Float16)f.x; p.h[1] = (_Float16)f.y;
    p.h[2] = (_Float16)f.z; p.h[3] = (_Float16)f.w;
    ((uint2*)d)[i] = p.u;
  }
}

// ---------------------------------------------------------------- GEMM
// y[r,f] = scale * ( sum_k A[r,k]*W[f,k] + bias[f] ),  one wave = 64x64 tile.
// MODE 0: f32 row-major out [M,N]
// MODE 1: f16 out to [B,H,S,D]   (Q and K)
// MODE 2: f16 out to [B,H,D,S]   (V transposed; feeds PV A-frags)
template <int MODE>
__global__ void gemm_wmma_64x64(const _Float16* __restrict__ A,
                                const _Float16* __restrict__ W,
                                const float* __restrict__ bias,
                                void* __restrict__ outp,
                                int M, int N, int K, float scale) {
  const int lane = threadIdx.x & 31;
  const int l15  = lane & 15;
  const int hi   = lane >> 4;
  const int akb  = hi ? 8 : 0;    // A-frag K base (ISA 16-bit A layout)
  const int bkb  = hi ? 8 : 0;    // B-frag K base
  const int m0 = blockIdx.y * 64;
  const int n0 = blockIdx.x * 64;

  v8f c[4][4];
#pragma unroll
  for (int i = 0; i < 4; ++i)
#pragma unroll
    for (int j = 0; j < 4; ++j) c[i][j] = zero8();

  for (int k0 = 0; k0 < K; k0 += 32) {
    AF a[4], bf[4];
#pragma unroll
    for (int mi = 0; mi < 4; ++mi) {
      const _Float16* ap = A + (size_t)(m0 + mi * 16 + l15) * K + k0 + akb;
      a[mi].h[0] = *(const v8h*)ap;
      a[mi].h[1] = *(const v8h*)(ap + 16);
    }
#pragma unroll
    for (int ni = 0; ni < 4; ++ni) {
      const _Float16* bp = W + (size_t)(n0 + ni * 16 + l15) * K + k0 + bkb;
      bf[ni].h[0] = *(const v8h*)bp;
      bf[ni].h[1] = *(const v8h*)(bp + 16);
    }
#pragma unroll
    for (int mi = 0; mi < 4; ++mi)
#pragma unroll
      for (int ni = 0; ni < 4; ++ni)
        c[mi][ni] = wmma_ff(a[mi], bf[ni], c[mi][ni]);
  }

  float bv[4];
#pragma unroll
  for (int ni = 0; ni < 4; ++ni) bv[ni] = bias[n0 + ni * 16 + l15];

#pragma unroll
  for (int mi = 0; mi < 4; ++mi) {
#pragma unroll
    for (int v = 0; v < 8; ++v) {
      const int r = m0 + mi * 16 + v + 8 * hi;   // = s*B + b
#pragma unroll
      for (int ni = 0; ni < 4; ++ni) {
        const int f = n0 + ni * 16 + l15;        // = h*D + d
        const float val = (c[mi][ni][v] + bv[ni]) * scale;
        if (MODE == 0) {
          ((float*)outp)[(size_t)r * N + f] = val;
        } else {
          const int s = r >> 2, b = r & 3;
          const int hh = f >> 6, d = f & 63;
          if (MODE == 1)
            ((_Float16*)outp)[(((size_t)b * H_ + hh) * S_ + s) * D_ + d] = (_Float16)val;
          else
            ((_Float16*)outp)[(((size_t)b * H_ + hh) * D_ + d) * S_ + s] = (_Float16)val;
        }
      }
    }
  }
}

// ---------------------------------------------------------------- attention fwd
// One wave per (b, h, 16-query block). Scores computed transposed:
//   scT[t,s] = sum_d K[t,d] * Q[s,d]    (A-frag = K rows, B-frag = Q rows)
// so lane owns query row s = lane&15 and 16 of 32 key positions (t = v+8*hi).
// Output accumulated transposed: oT[d,s] += sum_t V^T[d,t] * P^T[t,s].
__global__ void attn_fwd(const _Float16* __restrict__ Qh, const _Float16* __restrict__ Kh,
                         const _Float16* __restrict__ Vt, _Float16* __restrict__ ctx,
                         float* __restrict__ Mbuf, float* __restrict__ Lbuf) {
  __shared__ _Float16 lds_p[16][40];  // P^T staged as [s][t_local], padded rows
  const int lane = threadIdx.x & 31;
  const int l15 = lane & 15, hi = lane >> 4;
  const int akb = hi ? 8 : 0, bkb = hi ? 8 : 0;
  const int s0 = blockIdx.x * 16;
  const int h = blockIdx.y, b = blockIdx.z;
  const int srow = s0 + l15;          // this lane's query row

  const _Float16* Qp = Qh + ((size_t)b * H_ + h) * (size_t)S_ * D_;
  const _Float16* Kp = Kh + ((size_t)b * H_ + h) * (size_t)S_ * D_;
  const _Float16* Vp = Vt + ((size_t)b * H_ + h) * (size_t)D_ * S_;

  AF qb0, qb1;  // Q as B-frag: n = s (lane&15), k = d (chunks 0..31 / 32..63)
  {
    const _Float16* qr = Qp + (size_t)srow * D_;
    qb0.h[0] = *(const v8h*)(qr + bkb);      qb0.h[1] = *(const v8h*)(qr + bkb + 16);
    qb1.h[0] = *(const v8h*)(qr + 32 + bkb); qb1.h[1] = *(const v8h*)(qr + 48 + bkb);
  }

  float mlane = -__builtin_inff(), llane = 0.f;
  v8f o0 = zero8(), o1 = zero8(), o2 = zero8(), o3 = zero8();

  for (int t0 = 0; t0 < s0 + 16; t0 += 32) {
    const _Float16* kr0 = Kp + (size_t)(t0 + l15) * D_;
    const _Float16* kr1 = kr0 + 16 * D_;
    if (t0 + 32 < s0 + 16) {  // prefetch next key block (-> global_prefetch_b8)
      __builtin_prefetch(kr0 + 32 * D_, 0, 3);
      __builtin_prefetch(kr1 + 32 * D_, 0, 3);
    }
    AF ka00, ka01, ka10, ka11;  // K as A-frag: rows t, k = d
    ka00.h[0] = *(const v8h*)(kr0 + akb);      ka00.h[1] = *(const v8h*)(kr0 + 16 + akb);
    ka01.h[0] = *(const v8h*)(kr0 + 32 + akb); ka01.h[1] = *(const v8h*)(kr0 + 48 + akb);
    ka10.h[0] = *(const v8h*)(kr1 + akb);      ka10.h[1] = *(const v8h*)(kr1 + 16 + akb);
    ka11.h[0] = *(const v8h*)(kr1 + 32 + akb); ka11.h[1] = *(const v8h*)(kr1 + 48 + akb);

    v8f scT0 = zero8(), scT1 = zero8();
    scT0 = wmma_ff(ka00, qb0, scT0); scT0 = wmma_ff(ka01, qb1, scT0);
    scT1 = wmma_ff(ka10, qb0, scT1); scT1 = wmma_ff(ka11, qb1, scT1);

    // causal mask: element (t, s=srow) invalid if t > srow. Uniform branch:
    // only diagonal-straddling blocks pay the compare/cndmask cost.
    float x0[8], x1[8];
    if (t0 + 31 > s0) {
#pragma unroll
      for (int v = 0; v < 8; ++v) {
        const int t_a = t0 + v + 8 * hi;
        const int t_b = t_a + 16;
        x0[v] = (t_a > srow) ? -__builtin_inff() : scT0[v];
        x1[v] = (t_b > srow) ? -__builtin_inff() : scT1[v];
      }
    } else {
#pragma unroll
      for (int v = 0; v < 8; ++v) { x0[v] = scT0[v]; x1[v] = scT1[v]; }
    }
    // row max: in-lane tree + one cross-half shuffle
    float mx = fmaxf(x0[0], x1[0]);
#pragma unroll
    for (int v = 1; v < 8; ++v) mx = fmaxf(mx, fmaxf(x0[v], x1[v]));
    mx = fmaxf(mx, __shfl_xor(mx, 16, 32));
    const float mnew = fmaxf(mlane, mx);
    const float corr = __expf(mlane - mnew);
    mlane = mnew;

    float sum = 0.f;
    v8h ph0, ph1;
#pragma unroll
    for (int v = 0; v < 8; ++v) {
      const float e0 = __expf(x0[v] - mnew);
      const float e1 = __expf(x1[v] - mnew);
      sum += e0 + e1;
      ph0[v] = (_Float16)e0;
      ph1[v] = (_Float16)e1;
    }
    sum += __shfl_xor(sum, 16, 32);
    llane = llane * corr + sum;

#pragma unroll
    for (int v = 0; v < 8; ++v) {
      o0[v] *= corr; o1[v] *= corr; o2[v] *= corr; o3[v] *= corr;
    }
    // stage P^T as [s][t_local]: this lane's chunks are contiguous in t
    *(v8h*)&lds_p[l15][8 * hi]      = ph0;
    *(v8h*)&lds_p[l15][16 + 8 * hi] = ph1;
    __syncthreads();
    AF pb;  // P^T as B-frag: n = s (lane&15), k = t_local
    pb.h[0] = *(const v8h*)&lds_p[l15][bkb];
    pb.h[1] = *(const v8h*)&lds_p[l15][bkb + 16];
#pragma unroll
    for (int nd = 0; nd < 4; ++nd) {
      const _Float16* vr = Vp + (size_t)(nd * 16 + l15) * S_ + t0;
      AF va;  // V^T as A-frag: rows d, k = t_local
      va.h[0] = *(const v8h*)(vr + akb);
      va.h[1] = *(const v8h*)(vr + 16 + akb);
      v8f* op = nd == 0 ? &o0 : nd == 1 ? &o1 : nd == 2 ? &o2 : &o3;
      *op = wmma_ff(va, pb, *op);
    }
    __syncthreads();
  }

  // epilogue: oT[d, s=srow]; d = nd*16 + v + 8*hi contiguous in v -> packed stores
  const float linv = 1.0f / llane;
  _Float16* cp = ctx + ((size_t)srow * B_ + b) * E_ + h * D_ + 8 * hi;
  v8h e0, e1, e2, e3;
#pragma unroll
  for (int v = 0; v < 8; ++v) {
    e0[v] = (_Float16)(o0[v] * linv);
    e1[v] = (_Float16)(o1[v] * linv);
    e2[v] = (_Float16)(o2[v] * linv);
    e3[v] = (_Float16)(o3[v] * linv);
  }
  *(v8h*)(cp)      = e0;
  *(v8h*)(cp + 16) = e1;
  *(v8h*)(cp + 32) = e2;
  *(v8h*)(cp + 48) = e3;
  if (lane < 16) {
    const size_t ri = ((size_t)b * H_ + h) * S_ + srow;
    Mbuf[ri] = mlane;
    Lbuf[ri] = llane;
  }
}

// ---------------------------------------------------------------- attn_map
// One 4-wave block per (b, 16-query block). Waves own interleaved key blocks,
// loop all H heads accumulating exp(s-m)/l/H in registers (transposed score
// layout -> lane owns row s, contiguous t), then contiguous float4 stores of
// the full [16,S] strip (causal zeros included; d_out is poisoned).
__global__ void attn_map_k(const _Float16* __restrict__ Qh, const _Float16* __restrict__ Kh,
                           const float* __restrict__ Mbuf, const float* __restrict__ Lbuf,
                           float* __restrict__ amap) {
  __shared__ _Float16 qlds[H_ * 16 * D_];  // 32KB: Q tile for all heads
  __shared__ float mlds[H_ * 16];
  __shared__ float llds[H_ * 16];
  const int tid = threadIdx.x;
  const int wave = tid >> 5;
  const int lane = tid & 31;
  const int l15 = lane & 15, hi = lane >> 4;
  const int akb = hi ? 8 : 0, bkb = hi ? 8 : 0;
  const int s0 = blockIdx.x * 16;
  const int b = blockIdx.y;
  const int srow = s0 + l15;

  for (int idx = tid; idx < H_ * 16 * 8; idx += 128) {
    const int h = idx >> 7;
    const int r = (idx >> 3) & 15;
    const int c8 = idx & 7;
    *(v8h*)&qlds[(h * 16 + r) * D_ + c8 * 8] =
        *(const v8h*)(Qh + (((size_t)b * H_ + h) * S_ + s0 + r) * D_ + c8 * 8);
  }
  for (int idx = tid; idx < H_ * 16; idx += 128) {
    const int h = idx >> 4, r = idx & 15;
    const size_t qi = ((size_t)b * H_ + h) * S_ + s0 + r;
    mlds[idx] = Mbuf[qi];
    llds[idx] = (1.0f / (float)H_) / Lbuf[qi];
  }
  __syncthreads();

  for (int t0 = wave * 32; t0 < S_; t0 += 128) {
    float* rowp = amap + ((size_t)b * S_ + srow) * S_ + t0 + 8 * hi;
    if (t0 <= s0 + 15) {
      const bool need_mask = (t0 + 31 > s0);  // uniform across the wave
      float a0[8], a1[8];
#pragma unroll
      for (int v = 0; v < 8; ++v) { a0[v] = 0.f; a1[v] = 0.f; }
      for (int h = 0; h < H_; ++h) {
        AF qb0, qb1;
        const _Float16* qr = &qlds[(h * 16 + l15) * D_];
        qb0.h[0] = *(const v8h*)(qr + bkb);      qb0.h[1] = *(const v8h*)(qr + bkb + 16);
        qb1.h[0] = *(const v8h*)(qr + 32 + bkb); qb1.h[1] = *(const v8h*)(qr + 48 + bkb);
        const _Float16* kr0 = Kh + (((size_t)b * H_ + h) * S_ + t0 + l15) * D_;
        const _Float16* kr1 = kr0 + 16 * D_;
        AF ka00, ka01, ka10, ka11;
        ka00.h[0] = *(const v8h*)(kr0 + akb);      ka00.h[1] = *(const v8h*)(kr0 + 16 + akb);
        ka01.h[0] = *(const v8h*)(kr0 + 32 + akb); ka01.h[1] = *(const v8h*)(kr0 + 48 + akb);
        ka10.h[0] = *(const v8h*)(kr1 + akb);      ka10.h[1] = *(const v8h*)(kr1 + 16 + akb);
        ka11.h[0] = *(const v8h*)(kr1 + 32 + akb); ka11.h[1] = *(const v8h*)(kr1 + 48 + akb);
        v8f scT0 = zero8(), scT1 = zero8();
        scT0 = wmma_ff(ka00, qb0, scT0); scT0 = wmma_ff(ka01, qb1, scT0);
        scT1 = wmma_ff(ka10, qb0, scT1); scT1 = wmma_ff(ka11, qb1, scT1);
        const float mr = mlds[h * 16 + l15];
        const float li = llds[h * 16 + l15];
        if (need_mask) {
#pragma unroll
          for (int v = 0; v < 8; ++v) {
            const int t_a = t0 + v + 8 * hi;
            const int t_b = t_a + 16;
            a0[v] += (t_a > srow) ? 0.f : __expf(scT0[v] - mr) * li;
            a1[v] += (t_b > srow) ? 0.f : __expf(scT1[v] - mr) * li;
          }
        } else {
#pragma unroll
          for (int v = 0; v < 8; ++v) {
            a0[v] += __expf(scT0[v] - mr) * li;
            a1[v] += __expf(scT1[v] - mr) * li;
          }
        }
      }
      float4 f;
      f.x = a0[0]; f.y = a0[1]; f.z = a0[2]; f.w = a0[3]; *(float4*)(rowp) = f;
      f.x = a0[4]; f.y = a0[5]; f.z = a0[6]; f.w = a0[7]; *(float4*)(rowp + 4) = f;
      f.x = a1[0]; f.y = a1[1]; f.z = a1[2]; f.w = a1[3]; *(float4*)(rowp + 16) = f;
      f.x = a1[4]; f.y = a1[5]; f.z = a1[6]; f.w = a1[7]; *(float4*)(rowp + 20) = f;
    } else {
      const float4 z = {0.f, 0.f, 0.f, 0.f};
      *(float4*)(rowp) = z;
      *(float4*)(rowp + 4) = z;
      *(float4*)(rowp + 16) = z;
      *(float4*)(rowp + 20) = z;
    }
  }
}

// ---------------------------------------------------------------- launch
extern "C" void kernel_launch(void* const* d_in, const int* in_sizes, int n_in,
                              void* d_out, int out_size, void* d_ws, size_t ws_size,
                              hipStream_t stream) {
  (void)in_sizes; (void)n_in; (void)out_size; (void)ws_size;
  const float* q_f  = (const float*)d_in[0];
  const float* k_f  = (const float*)d_in[1];
  const float* v_f  = (const float*)d_in[2];
  // d_in[3] = attn_mask (causal), applied analytically
  const float* Wq = (const float*)d_in[4];
  const float* bq = (const float*)d_in[5];
  const float* Wk = (const float*)d_in[6];
  const float* bk = (const float*)d_in[7];
  const float* Wv = (const float*)d_in[8];
  const float* bv_in = (const float*)d_in[9];
  const float* Wo = (const float*)d_in[10];
  const float* bo = (const float*)d_in[11];

  char* ws = (char*)d_ws;
  size_t off = 0;
  auto alloc = [&](size_t bytes) -> void* {
    void* p = ws + off;
    off += (bytes + 255) & ~(size_t)255;
    return p;
  };
  _Float16* xq = (_Float16*)alloc((size_t)MROWS * E_ * 2);
  _Float16* xk = (_Float16*)alloc((size_t)MROWS * E_ * 2);
  _Float16* xv = (_Float16*)alloc((size_t)MROWS * E_ * 2);
  _Float16* wqh = (_Float16*)alloc((size_t)E_ * E_ * 2);
  _Float16* wkh = (_Float16*)alloc((size_t)E_ * E_ * 2);
  _Float16* wvh = (_Float16*)alloc((size_t)E_ * E_ * 2);
  _Float16* woh = (_Float16*)alloc((size_t)E_ * E_ * 2);
  _Float16* Qh = (_Float16*)alloc((size_t)MROWS * E_ * 2);  // [B,H,S,D]
  _Float16* Kh = (_Float16*)alloc((size_t)MROWS * E_ * 2);  // [B,H,S,D]
  _Float16* Vt = (_Float16*)alloc((size_t)MROWS * E_ * 2);  // [B,H,D,S]
  float* Mbuf = (float*)alloc((size_t)B_ * H_ * S_ * 4);
  float* Lbuf = (float*)alloc((size_t)B_ * H_ * S_ * 4);
  _Float16* ctx = xq;  // xq dead after Q projection; reuse for context [S,B,E]

  const int n4x = MROWS * E_ / 4, n4w = E_ * E_ / 4;
  cvt_f32_f16_x4<<<dim3((n4x + 255) / 256), dim3(256), 0, stream>>>(q_f, xq, n4x);
  cvt_f32_f16_x4<<<dim3((n4x + 255) / 256), dim3(256), 0, stream>>>(k_f, xk, n4x);
  cvt_f32_f16_x4<<<dim3((n4x + 255) / 256), dim3(256), 0, stream>>>(v_f, xv, n4x);
  cvt_f32_f16_x4<<<dim3((n4w + 255) / 256), dim3(256), 0, stream>>>(Wq, wqh, n4w);
  cvt_f32_f16_x4<<<dim3((n4w + 255) / 256), dim3(256), 0, stream>>>(Wk, wkh, n4w);
  cvt_f32_f16_x4<<<dim3((n4w + 255) / 256), dim3(256), 0, stream>>>(Wv, wvh, n4w);
  cvt_f32_f16_x4<<<dim3((n4w + 255) / 256), dim3(256), 0, stream>>>(Wo, woh, n4w);

  const dim3 ggrid(E_ / 64, MROWS / 64);  // (16,128)
  gemm_wmma_64x64<1><<<ggrid, dim3(32), 0, stream>>>(xq, wqh, bq, Qh, MROWS, E_, E_, 0.125f);
  gemm_wmma_64x64<1><<<ggrid, dim3(32), 0, stream>>>(xk, wkh, bk, Kh, MROWS, E_, E_, 1.0f);
  gemm_wmma_64x64<2><<<ggrid, dim3(32), 0, stream>>>(xv, wvh, bv_in, Vt, MROWS, E_, E_, 1.0f);

  attn_fwd<<<dim3(S_ / 16, H_, B_), dim3(32), 0, stream>>>(Qh, Kh, Vt, ctx, Mbuf, Lbuf);

  float* amap = (float*)d_out + (size_t)MROWS * E_;
  attn_map_k<<<dim3(S_ / 16, B_), dim3(128), 0, stream>>>(Qh, Kh, Mbuf, Lbuf, amap);

  gemm_wmma_64x64<0><<<ggrid, dim3(32), 0, stream>>>(ctx, woh, bo, d_out, MROWS, E_, E_, 1.0f);
}
